// LinearAttention_49417893707931
// MI455X (gfx1250) — matrix operand order repaired and checked by reading, not compiled, not run on hardware
//
#include <hip/hip_runtime.h>

#define B_  2
#define L_  4096
#define DM_ 1024
#define H_  16
#define D_  64

#define AS1 __attribute__((address_space(1)))
#define AS3 __attribute__((address_space(3)))

#if defined(__AMDGCN__) && __has_builtin(__builtin_amdgcn_global_load_async_to_lds_b128) && __has_builtin(__builtin_amdgcn_s_wait_asynccnt)
#define HAVE_ASYNC_LDS 1
#else
#define HAVE_ASYNC_LDS 0
#endif

typedef int v4i __attribute__((vector_size(16)));   // matches builtin param type

typedef __attribute__((ext_vector_type(16))) __bf16 v16bf;
typedef __attribute__((ext_vector_type(8)))  float  v8f;

union FragBF { v16bf v; unsigned u[8]; };

__device__ __forceinline__ unsigned short f2bf(float f) {
  union { float f; unsigned u; } x; x.f = f;
  unsigned r = x.u + 0x7FFFu + ((x.u >> 16) & 1u);   // round-to-nearest-even
  return (unsigned short)(r >> 16);
}
__device__ __forceinline__ float bf2f(unsigned short s) {
  union { unsigned u; float f; } x; x.u = ((unsigned)s) << 16;
  return x.f;
}

// A-fragment (16x32 bf16, MxK) from M-major storage (stride in elements).
// ISA layout: lanes 0-15 (hl=0) hold K {0-7,16-23}; lanes 16-31 hold K {8-15,24-31}.
__device__ __forceinline__ v16bf frag_a(const unsigned short* t, int stride, int m, int hl, int koff) {
  FragBF f;
  const unsigned short* rp = t + (size_t)m * stride + koff + hl * 8;
#pragma unroll
  for (int v = 0; v < 8; ++v)
    f.u[v] = *reinterpret_cast<const unsigned*>(rp + (((v & 4) << 2) + ((v & 3) << 1)));
  return f.v;
}

// B-fragment (32x16 bf16, KxN) read from N-major (transposed) storage.
// ISA layout: N = lane&15; lanes 0-15 (hl=0) hold K {0-7,16-23}; lanes 16-31 hold K {8-15,24-31}.
__device__ __forceinline__ v16bf frag_b(const unsigned short* t, int stride, int n, int hl, int koff) {
  FragBF f;
  const unsigned short* rp = t + (size_t)n * stride + koff + hl * 8;
#pragma unroll
  for (int v = 0; v < 8; ++v)
    f.u[v] = *reinterpret_cast<const unsigned*>(rp + (((v & 4) << 2) + ((v & 3) << 1)));
  return f.v;
}

__device__ __forceinline__ void load4bf(const float* p, ushort4& o) {
  const float4 f = *reinterpret_cast<const float4*>(p);
  o.x = f2bf(f.x); o.y = f2bf(f.y); o.z = f2bf(f.z); o.w = f2bf(f.w);
}
__device__ __forceinline__ void load4bf(const unsigned short* p, ushort4& o) {
  o = *reinterpret_cast<const ushort4*>(p);
}

// ---------------------------------------------------------------------------
// GEMM: C[M,N] = act(A[M,K] * W[K,N] + bias)
// MODE 0: fp32 out (final proj); MODE 1: bf16 out; MODE 2: bf16 out + elu+1
// ---------------------------------------------------------------------------
#define BM 128
#define BN 128
#define BK 32
#define LDT 40   // padded K-stride (elements); 80B rows keep 16B chunks aligned

template<int MODE, typename TA>
__global__ __launch_bounds__(256) void gemm_kernel(
    const TA* __restrict__ A, const float* __restrict__ W,
    const float* __restrict__ bias,
    unsigned short* __restrict__ Obf, float* __restrict__ Of,
    int M, int N, int K) {
  __shared__ unsigned short As[2][BM * LDT];
  __shared__ unsigned short Bt[2][BN * LDT];   // W tile stored N-major (transposed)

  const int tid  = threadIdx.x;
  const int row0 = blockIdx.y * BM;
  const int col0 = blockIdx.x * BN;
  const int lane = tid & 31, wid = tid >> 5;
  const int hl = lane >> 4, ln = lane & 15;
  const int wm = (wid >> 2) * 64;   // 2 waves along M
  const int wn = (wid & 3) * 32;    // 4 waves along N

  v8f acc[4][2] = {};

  auto load_tiles = [&](int k0, int buf) {
#if HAVE_ASYNC_LDS
    if constexpr (sizeof(TA) == 2) {
      // bf16 A needs no conversion: DMA global -> LDS, tracked by ASYNCcnt.
#pragma unroll
      for (int i = 0; i < 2; ++i) {
        int idx = tid + i * 256;           // 512 x 16B chunks (128x32 bf16)
        int r = idx >> 2, c8 = (idx & 3) << 3;
        __builtin_amdgcn_global_load_async_to_lds_b128(
            (AS1 v4i*)(unsigned long long)(const void*)(A + (size_t)(row0 + r) * K + k0 + c8),
            (AS3 v4i*)&As[buf][r * LDT + c8], 0, 0);
      }
    } else
#endif
    {
#pragma unroll
      for (int i = 0; i < 4; ++i) {            // A: 128x32 (1024 x vec4)
        int idx = tid + i * 256;
        int r = idx >> 3, c4 = (idx & 7) << 2;
        ushort4 pk;
        load4bf(A + (size_t)(row0 + r) * K + k0 + c4, pk);
        *reinterpret_cast<ushort4*>(&As[buf][r * LDT + c4]) = pk;
      }
    }
#pragma unroll
    for (int i = 0; i < 4; ++i) {              // W: 32x128, transposed store
      int idx = tid + i * 256;
      int kk = idx >> 5, n4 = (idx & 31) << 2;
      const float4 f = *reinterpret_cast<const float4*>(W + (size_t)(k0 + kk) * N + col0 + n4);
      Bt[buf][(n4 + 0) * LDT + kk] = f2bf(f.x);
      Bt[buf][(n4 + 1) * LDT + kk] = f2bf(f.y);
      Bt[buf][(n4 + 2) * LDT + kk] = f2bf(f.z);
      Bt[buf][(n4 + 3) * LDT + kk] = f2bf(f.w);
    }
  };

  load_tiles(0, 0);
#if HAVE_ASYNC_LDS
  if constexpr (sizeof(TA) == 2) __builtin_amdgcn_s_wait_asynccnt(0);
#endif
  __syncthreads();

  const int KT = K / BK;
  int buf = 0;
  for (int kt = 0; kt < KT; ++kt) {
    if (kt + 1 < KT) {
      __builtin_prefetch(W + (size_t)((kt + 1) * BK) * N + col0 + (tid & (BN - 1)), 0, 1);
      load_tiles((kt + 1) * BK, buf ^ 1);
    }
    v16bf af[4], bfr[2];
#pragma unroll
    for (int mi = 0; mi < 4; ++mi)
      af[mi] = frag_a(&As[buf][(wm + mi * 16) * LDT], LDT, ln, hl, 0);
#pragma unroll
    for (int ni = 0; ni < 2; ++ni)
      bfr[ni] = frag_b(&Bt[buf][(wn + ni * 16) * LDT], LDT, ln, hl, 0);
#pragma unroll
    for (int mi = 0; mi < 4; ++mi)
#pragma unroll
      for (int ni = 0; ni < 2; ++ni)
        acc[mi][ni] = __builtin_amdgcn_wmma_f32_16x16x32_bf16(
            false, af[mi], false, bfr[ni], (short)0, acc[mi][ni], false, false);
#if HAVE_ASYNC_LDS
    if constexpr (sizeof(TA) == 2) __builtin_amdgcn_s_wait_asynccnt(0);
#endif
    __syncthreads();
    buf ^= 1;
  }

#pragma unroll
  for (int mi = 0; mi < 4; ++mi) {
#pragma unroll
    for (int ni = 0; ni < 2; ++ni) {
      const int col = col0 + wn + ni * 16 + ln;
      const float bv = bias[col];
#pragma unroll
      for (int r = 0; r < 8; ++r) {
        const int row = row0 + wm + mi * 16 + r + 8 * hl;  // C layout: lanes>=16 -> +8 rows
        float x = acc[mi][ni][r] + bv;
        if constexpr (MODE == 2) x = (x > 0.0f) ? (x + 1.0f) : __expf(x);  // elu(x)+1
        if constexpr (MODE == 0)
          Of[(size_t)row * N + col] = x;
        else
          Obf[(size_t)row * N + col] = f2bf(x);
      }
    }
  }
}

// ---------------------------------------------------------------------------
// denom[bh*L + l] = sum_d q[l,d]*k[l,d] + 1e-6   (feature-mapped q,k)
// ---------------------------------------------------------------------------
__global__ __launch_bounds__(256) void denom_kernel(
    const unsigned short* __restrict__ qbf, const unsigned short* __restrict__ kbf,
    float* __restrict__ denom) {
  const int t  = blockIdx.x * 256 + threadIdx.x;  // t = bh*L + l
  const int l  = t & (L_ - 1);
  const int bh = t >> 12;
  const int bb = bh >> 4, h = bh & 15;
  const size_t row = (size_t)(bb * L_ + l) * DM_ + h * D_;
  float s = 0.f;
#pragma unroll
  for (int d = 0; d < D_; d += 4) {
    ushort4 qv = *reinterpret_cast<const ushort4*>(qbf + row + d);
    ushort4 kv = *reinterpret_cast<const ushort4*>(kbf + row + d);
    s += bf2f(qv.x) * bf2f(kv.x) + bf2f(qv.y) * bf2f(kv.y)
       + bf2f(qv.z) * bf2f(kv.z) + bf2f(qv.w) * bf2f(kv.w);
  }
  denom[t] = s + 1e-6f;
}

// ---------------------------------------------------------------------------
// S[e,d] = sum_l K[l,e] * Q[l,d] per (b,h); store bf16 transposed St[d][e]
// One block per (b,h); 8 waves x 2 tiles cover the 64x64 output; K-reduce L.
// ---------------------------------------------------------------------------
__global__ __launch_bounds__(256) void s_kernel(
    const unsigned short* __restrict__ kbf, const unsigned short* __restrict__ qbf,
    unsigned short* __restrict__ St) {
  __shared__ unsigned short Kt[D_ * LDT];   // [e][l] e-major
  __shared__ unsigned short Qt[D_ * LDT];   // [d][l] d-major
  const int bh = blockIdx.x;
  const int bb = bh >> 4, h = bh & 15;
  const int tid = threadIdx.x;
  const int lane = tid & 31, wid = tid >> 5;
  const int hl = lane >> 4, ln = lane & 15;

  v8f acc[2] = {};

  for (int l0 = 0; l0 < L_; l0 += 32) {
    __syncthreads();
#pragma unroll
    for (int i = 0; i < 2; ++i) {          // 32l x 64c per matrix, transposed stage
      int idx = tid + i * 256;
      int l = idx >> 4, c4 = (idx & 15) << 2;
      size_t g = (size_t)(bb * L_ + l0 + l) * DM_ + h * D_ + c4;
      ushort4 kv = *reinterpret_cast<const ushort4*>(kbf + g);
      ushort4 qv = *reinterpret_cast<const ushort4*>(qbf + g);
      Kt[(c4 + 0) * LDT + l] = kv.x; Kt[(c4 + 1) * LDT + l] = kv.y;
      Kt[(c4 + 2) * LDT + l] = kv.z; Kt[(c4 + 3) * LDT + l] = kv.w;
      Qt[(c4 + 0) * LDT + l] = qv.x; Qt[(c4 + 1) * LDT + l] = qv.y;
      Qt[(c4 + 2) * LDT + l] = qv.z; Qt[(c4 + 3) * LDT + l] = qv.w;
    }
    __syncthreads();
#pragma unroll
    for (int j = 0; j < 2; ++j) {
      int t = wid * 2 + j, te = t >> 2, td = t & 3;
      v16bf a = frag_a(&Kt[te * 16 * LDT], LDT, ln, hl, 0);  // A[m=e][k=l]
      v16bf b = frag_b(&Qt[td * 16 * LDT], LDT, ln, hl, 0);  // B[k=l][n=d]
      acc[j] = __builtin_amdgcn_wmma_f32_16x16x32_bf16(
          false, a, false, b, (short)0, acc[j], false, false);
    }
  }
#pragma unroll
  for (int j = 0; j < 2; ++j) {
    int t = wid * 2 + j, te = t >> 2, td = t & 3;
#pragma unroll
    for (int r = 0; r < 8; ++r)   // element (e = te*16+r+8*hl, d = td*16+ln) -> St[d][e]
      St[(size_t)bh * (D_ * D_) + (td * 16 + ln) * D_ + te * 16 + r + 8 * hl]
          = f2bf(acc[j][r]);
  }
}

// ---------------------------------------------------------------------------
// ctx[l, h*64+d] = (sum_e v[l,e] * S[e,d]) / denom[l]
// Grid: (L/128, B*H). 8 waves x 16 rows each; S resident in LDS.
// ---------------------------------------------------------------------------
#define LDS_S 72
__global__ __launch_bounds__(256) void ctx_kernel(
    const unsigned short* __restrict__ vbf, const unsigned short* __restrict__ St,
    const float* __restrict__ denom, unsigned short* __restrict__ ctx) {
  __shared__ unsigned short Ssh[D_ * LDS_S];   // [d][e]
  const int bh = blockIdx.y;
  const int bb = bh >> 4, h = bh & 15;
  const int l0 = blockIdx.x * 128;
  const int tid = threadIdx.x;
  const int lane = tid & 31, wid = tid >> 5;
  const int hl = lane >> 4, ln = lane & 15;

#if HAVE_ASYNC_LDS
#pragma unroll
  for (int i = 0; i < 2; ++i) {
    int idx = tid + i * 256;                 // 512 x 16B chunks (64x64 bf16)
    int d = idx >> 2, e8 = (idx & 3) << 3;
    __builtin_amdgcn_global_load_async_to_lds_b128(
        (AS1 v4i*)(unsigned long long)(const void*)(St + (size_t)bh * (D_ * D_) + d * D_ + e8),
        (AS3 v4i*)&Ssh[d * LDS_S + e8], 0, 0);
  }
  __builtin_amdgcn_s_wait_asynccnt(0);
#else
#pragma unroll
  for (int i = 0; i < 4; ++i) {
    int idx = tid + i * 256;
    int d = idx >> 4, e4 = (idx & 15) << 2;
    ushort4 sv = *reinterpret_cast<const ushort4*>(St + (size_t)bh * (D_ * D_) + d * D_ + e4);
    *reinterpret_cast<ushort4*>(&Ssh[d * LDS_S + e4]) = sv;
  }
#endif
  __syncthreads();

  // A fragments straight from global: v rows are contiguous in e.
  const unsigned short* arow = vbf + (size_t)(bb * L_ + l0 + wid * 16) * DM_ + h * D_;
  v16bf a[2];
#pragma unroll
  for (int kc = 0; kc < 2; ++kc)
    a[kc] = frag_a(arow, DM_, ln, hl, kc * 32);

  v8f acc[4] = {};
#pragma unroll
  for (int ni = 0; ni < 4; ++ni)
#pragma unroll
    for (int kc = 0; kc < 2; ++kc) {
      v16bf b = frag_b(&Ssh[ni * 16 * LDS_S], LDS_S, ln, hl, kc * 32);
      acc[ni] = __builtin_amdgcn_wmma_f32_16x16x32_bf16(
          false, a[kc], false, b, (short)0, acc[ni], false, false);
    }

  float dinv[8];
#pragma unroll
  for (int r = 0; r < 8; ++r)
    dinv[r] = 1.0f / denom[(size_t)bh * L_ + l0 + wid * 16 + r + 8 * hl];

#pragma unroll
  for (int ni = 0; ni < 4; ++ni) {
    const int col = h * D_ + ni * 16 + ln;
#pragma unroll
    for (int r = 0; r < 8; ++r) {
      const size_t row = (size_t)(bb * L_ + l0 + wid * 16 + r + 8 * hl);
      ctx[row * DM_ + col] = f2bf(acc[ni][r] * dinv[r]);
    }
  }
}

// ---------------------------------------------------------------------------
extern "C" void kernel_launch(void* const* d_in, const int* in_sizes, int n_in,
                              void* d_out, int out_size, void* d_ws, size_t ws_size,
                              hipStream_t stream) {
  const float* query = (const float*)d_in[0];
  const float* key   = (const float*)d_in[1];
  const float* value = (const float*)d_in[2];
  const float* Wq = (const float*)d_in[3];  const float* bq = (const float*)d_in[4];
  const float* Wk = (const float*)d_in[5];  const float* bk = (const float*)d_in[6];
  const float* Wv = (const float*)d_in[7];  const float* bv = (const float*)d_in[8];
  const float* Wo = (const float*)d_in[9];  const float* bo = (const float*)d_in[10];

  const size_t MAT = (size_t)B_ * L_ * DM_;        // 8M elements
  unsigned short* qbf  = (unsigned short*)d_ws;
  unsigned short* kbf  = qbf + MAT;
  unsigned short* vbf  = kbf + MAT;
  unsigned short* ctx  = vbf + MAT;
  unsigned short* St   = ctx + MAT;                 // B*H * 64*64 bf16
  float*          den  = (float*)(St + (size_t)B_ * H_ * D_ * D_);

  const int M = B_ * L_, N = DM_, K = DM_;
  dim3 blk(256);
  dim3 grid(N / BN, M / BM);   // (8, 64)

  gemm_kernel<2, float><<<grid, blk, 0, stream>>>(query, Wq, bq, qbf, nullptr, M, N, K);
  gemm_kernel<2, float><<<grid, blk, 0, stream>>>(key,   Wk, bk, kbf, nullptr, M, N, K);
  gemm_kernel<1, float><<<grid, blk, 0, stream>>>(value, Wv, bv, vbf, nullptr, M, N, K);

  denom_kernel<<<(B_ * H_ * L_) / 256, blk, 0, stream>>>(qbf, kbf, den);
  s_kernel<<<B_ * H_, blk, 0, stream>>>(kbf, qbf, St);
  ctx_kernel<<<dim3(L_ / 128, B_ * H_), blk, 0, stream>>>(vbf, St, den, ctx);

  gemm_kernel<0, unsigned short><<<grid, blk, 0, stream>>>(ctx, Wo, bo, nullptr,
                                                           (float*)d_out, M, N, K);
}
